// InnerSelfAttention_41412074668336
// MI455X (gfx1250) — matrix operand order repaired and checked by reading, not compiled, not run on hardware
//
#include <hip/hip_runtime.h>

#define H_HEADS 16
#define HEAD_D  64
#define WIN     256
#define SEQ     2048
#define BATCH   2
#define EMB     1024
#define MTOT    (BATCH * SEQ)
#define NKMAX   288   // 17 key tiles (272) padded to a multiple of 32

typedef __bf16 bf16_t;
typedef bf16_t v16bf __attribute__((ext_vector_type(16)));
typedef bf16_t v8bf  __attribute__((ext_vector_type(8)));
typedef bf16_t v4bf  __attribute__((ext_vector_type(4)));
typedef float  v8f   __attribute__((ext_vector_type(8)));

#if __has_builtin(__builtin_amdgcn_global_load_async_to_lds_b128) && \
    __has_builtin(__builtin_amdgcn_s_wait_asynccnt)
#define USE_ASYNC 1
typedef int vi4 __attribute__((ext_vector_type(4)));
typedef __attribute__((address_space(1))) vi4 as1_vi4;   // global
typedef __attribute__((address_space(3))) vi4 as3_vi4;   // LDS
#else
#define USE_ASYNC 0
#endif

// Copy one 16-byte chunk global -> LDS (async engine when available).
static __device__ __forceinline__ void stage_chunk(const bf16_t* g, bf16_t* l) {
#if USE_ASYNC
  __builtin_amdgcn_global_load_async_to_lds_b128(
      (as1_vi4*)(uintptr_t)g, (as3_vi4*)(uintptr_t)l, 0, 0);
#else
  *(v8bf*)l = *(const v8bf*)g;
#endif
}

static __device__ __forceinline__ void async_wait_all() {
#if USE_ASYNC
  __builtin_amdgcn_s_wait_asynccnt(0);
#endif
}

static __device__ __forceinline__ v16bf concat8(v8bf x, v8bf y) {
  return __builtin_shufflevector(x, y, 0,1,2,3,4,5,6,7,8,9,10,11,12,13,14,15);
}

// A fragment: 16x32 bf16 (MxK). Source row-major, rows `stride` elements apart.
// Lane L holds row (L&15); K halves {0..7,16..23} or {8..15,24..31} by L>>4.
static __device__ __forceinline__ v16bf load_a_frag(const bf16_t* base, int stride, int lane) {
  const int r  = lane & 15;
  const int hi = lane >> 4;
  const bf16_t* p = base + (size_t)r * stride + hi * 8;
  v8bf x = *(const v8bf*)p;
  v8bf y = *(const v8bf*)(p + 16);
  return concat8(x, y);
}

// B fragment: 32x16 bf16 (KxN). Source holds B^T row-major ([N][K]): row n of the
// source is column n of B. Lane n (resp. n+16) holds K {0..7,16..23} (resp. {8..15,24..31}) of col n.
static __device__ __forceinline__ v16bf load_b_frag(const bf16_t* base, int stride, int lane) {
  const int n  = lane & 15;
  const int hi = lane >> 4;
  const bf16_t* p = base + (size_t)n * stride + hi * 8;
  v8bf x = *(const v8bf*)p;
  v8bf y = *(const v8bf*)(p + 16);
  return concat8(x, y);
}

static __device__ __forceinline__ v8f wmma_bf16(v16bf a, v16bf b, v8f c) {
  return __builtin_amdgcn_wmma_f32_16x16x32_bf16(false, a, false, b, (short)0, c, false, false);
}

// ---------------------------------------------------------------- fp32 -> bf16
__global__ void cvt_f32_to_bf16(const float* __restrict__ in, bf16_t* __restrict__ out, int n4) {
  int i = blockIdx.x * blockDim.x + threadIdx.x;
  const int stride = gridDim.x * blockDim.x;
  const float4* in4 = (const float4*)in;
  v4bf* out4 = (v4bf*)out;
  for (; i < n4; i += stride) {
    float4 f = in4[i];
    v4bf b;
    b[0] = (bf16_t)f.x; b[1] = (bf16_t)f.y; b[2] = (bf16_t)f.z; b[3] = (bf16_t)f.w;
    out4[i] = b;
  }
}

// ---------------------------------------------------------------- GEMM y = A @ W^T
// A: [MTOT][EMB] bf16 row-major, W: [EMB][EMB] bf16 row-major ([N][K]).
// Block: 8 waves, 128x128 tile; wave: 64x32 (4x2 accumulators of 16x16).
// Double-buffered LDS staging of 128x64 A/B tiles via async-to-LDS.
#define EPI_BHSD 0   // scatter bf16 to [B,H,S,D]  (Q, K)
#define EPI_BHDS 2   // scatter bf16 to [B,H,D,S]  (V transposed)
#define EPI_F32B 3   // fp32 out + bias            (final projection)

#define KSTEP 64
#define LSTR  (KSTEP + 8)   // padded LDS row stride (elements)

__global__ __launch_bounds__(256) void gemm_x_wt(
    const bf16_t* __restrict__ A, const bf16_t* __restrict__ W,
    void* __restrict__ out, const float* __restrict__ bias, int mode)
{
  __shared__ __attribute__((aligned(16))) bf16_t sA[2][128][LSTR];
  __shared__ __attribute__((aligned(16))) bf16_t sB[2][128][LSTR];

  const int tid  = threadIdx.x;
  const int lane = tid & 31;
  const int wave = tid >> 5;
  const int mw = (wave >> 2) * 64;   // wave row offset inside the 128-tile
  const int nw = (wave & 3) * 32;    // wave col offset inside the 128-tile
  const int m0 = blockIdx.y * 128;
  const int n0 = blockIdx.x * 128;

  // Staging map: 1024 16-byte chunks per 128x64 tile; 4 chunks/thread each for A and B.
  const int cbase = tid * 4;

  auto stage = [&](int buf, int k0) {
    #pragma unroll
    for (int q = 0; q < 4; ++q) {
      const int c   = cbase + q;
      const int row = c >> 3;          // 0..127
      const int off = (c & 7) * 8;     // 0..56 elements
      stage_chunk(A + (size_t)(m0 + row) * EMB + k0 + off, &sA[buf][row][off]);
      stage_chunk(W + (size_t)(n0 + row) * EMB + k0 + off, &sB[buf][row][off]);
    }
  };

  const v8f vzero = {0.f,0.f,0.f,0.f,0.f,0.f,0.f,0.f};
  v8f acc[4][2];
  #pragma unroll
  for (int i = 0; i < 4; ++i)
    #pragma unroll
    for (int j = 0; j < 2; ++j) acc[i][j] = vzero;

  stage(0, 0);
  async_wait_all();
  __syncthreads();

  for (int k0 = 0; k0 < EMB; k0 += KSTEP) {
    const int buf = (k0 / KSTEP) & 1;
    const bool more = (k0 + KSTEP) < EMB;
    if (more) stage(buf ^ 1, k0 + KSTEP);   // prefetch next stage while computing

    #pragma unroll
    for (int kc = 0; kc < KSTEP; kc += 32) {
      v16bf a[4], b[2];
      #pragma unroll
      for (int i = 0; i < 4; ++i)
        a[i] = load_a_frag(&sA[buf][mw + 16*i][kc], LSTR, lane);
      #pragma unroll
      for (int j = 0; j < 2; ++j)
        b[j] = load_b_frag(&sB[buf][nw + 16*j][kc], LSTR, lane);
      #pragma unroll
      for (int i = 0; i < 4; ++i)
        #pragma unroll
        for (int j = 0; j < 2; ++j)
          acc[i][j] = wmma_bf16(a[i], b[j], acc[i][j]);
    }

    if (more) {
      async_wait_all();
      __syncthreads();
    }
  }

  const int col16 = lane & 15;
  const int rb    = (lane >> 4) * 8;
  #pragma unroll
  for (int i = 0; i < 4; ++i) {
    #pragma unroll
    for (int j = 0; j < 2; ++j) {
      const int col = n0 + nw + 16*j + col16;
      #pragma unroll
      for (int r = 0; r < 8; ++r) {
        const int m = m0 + mw + 16*i + rb + r;
        const float v = acc[i][j][r];
        if (mode == EPI_F32B) {
          ((float*)out)[(size_t)m * EMB + col] = v + bias[col];
        } else {
          const int bb = m >> 11, s = m & (SEQ - 1);
          const int hh = col >> 6, d = col & (HEAD_D - 1);
          const size_t idx = (mode == EPI_BHSD)
            ? (((size_t)(bb * H_HEADS + hh) * SEQ + s) * HEAD_D + d)
            : (((size_t)(bb * H_HEADS + hh) * HEAD_D + d) * SEQ + s);
          ((bf16_t*)out)[idx] = (bf16_t)v;
        }
      }
    }
  }
}

// ---------------------------------------------------------------- banded attention
// One wave per 16-query tile per (b,h). Keys j in (i-WIN, i] -> <= 17 key tiles.
__global__ __launch_bounds__(32) void attn_sliding_window(
    const bf16_t* __restrict__ Q,  // [BH][S][D]
    const bf16_t* __restrict__ Km, // [BH][S][D]
    const bf16_t* __restrict__ Vt, // [BH][D][S] (padded)
    bf16_t* __restrict__ Ctx)      // [B][S][E] merged heads
{
  __shared__ __attribute__((aligned(16))) float  sS[16][NKMAX];
  __shared__ __attribute__((aligned(16))) bf16_t sP[16][NKMAX];

  const int lane = threadIdx.x;
  const int bh = blockIdx.y;
  const int b  = bh >> 4;
  const int h  = bh & 15;
  const int q0 = blockIdx.x * 16;

  const int kstart = (q0 >= WIN) ? (q0 - WIN) : 0;   // multiple of 16
  const int nkt = (q0 + 16 - kstart) >> 4;           // <= 17
  const int NC  = nkt * 16;
  const int NCp = ((NC + 31) >> 5) << 5;             // pad to K=32 chunks

  const bf16_t* Qb = Q  + (size_t)bh * SEQ * HEAD_D;
  const bf16_t* Kb = Km + (size_t)bh * SEQ * HEAD_D;
  const bf16_t* Vb = Vt + (size_t)bh * HEAD_D * SEQ;

  const v16bf qa0 = load_a_frag(Qb + (size_t)q0 * HEAD_D +  0, HEAD_D, lane);
  const v16bf qa1 = load_a_frag(Qb + (size_t)q0 * HEAD_D + 32, HEAD_D, lane);

  const v8f vzero = {0.f,0.f,0.f,0.f,0.f,0.f,0.f,0.f};
  const int col16 = lane & 15;
  const int rb    = (lane >> 4) * 8;

  // scores = Q K^T (no 1/sqrt(d) scale, faithful to reference), band-masked into LDS
  for (int t = 0; t < nkt; ++t) {
    const int jb = kstart + t * 16;
    v16bf kb0 = load_b_frag(Kb + (size_t)jb * HEAD_D +  0, HEAD_D, lane);
    v16bf kb1 = load_b_frag(Kb + (size_t)jb * HEAD_D + 32, HEAD_D, lane);
    v8f c = vzero;
    c = wmma_bf16(qa0, kb0, c);
    c = wmma_bf16(qa1, kb1, c);
    const int j = jb + col16;
    #pragma unroll
    for (int r = 0; r < 8; ++r) {
      const int i = q0 + rb + r;
      sS[rb + r][t * 16 + col16] = ((j <= i) && (j > i - WIN)) ? c[r] : -1e30f;
    }
  }
  __syncthreads();

  // softmax: lane (row = lane&15, half = lane>>4), combine halves with shfl_xor(16)
  {
    const int row   = lane & 15;
    const int half  = NCp >> 1;
    const int c_lo  = (lane >> 4) * half;
    const int c_hi0 = c_lo + half;
    const int c_hi  = (c_hi0 < NC) ? c_hi0 : NC;
    float mx = -1e30f;
    for (int c = c_lo; c < c_hi; ++c) mx = fmaxf(mx, sS[row][c]);
    mx = fmaxf(mx, __shfl_xor(mx, 16, 32));
    float sum = 0.f;
    for (int c = c_lo; c < c_hi; ++c) {
      float e = __expf(sS[row][c] - mx);
      sS[row][c] = e;
      sum += e;
    }
    sum += __shfl_xor(sum, 16, 32);
    const float inv = 1.f / sum;
    for (int c = c_lo; c < c_hi; ++c) sP[row][c] = (bf16_t)(sS[row][c] * inv);
    for (int c = c_hi; c < c_hi0; ++c) sP[row][c] = (bf16_t)0.f;  // zero pad
  }
  __syncthreads();

  // ctx = P @ V : A-frags from LDS probs, B-frags contiguous from V^T rows
  v8f acc[4] = {vzero, vzero, vzero, vzero};
  for (int c0 = 0; c0 < NCp; c0 += 32) {
    v16bf pa = load_a_frag(&sP[0][0] + c0, NKMAX, lane);
    #pragma unroll
    for (int dt = 0; dt < 4; ++dt) {
      v16bf vb = load_b_frag(Vb + (size_t)(dt * 16) * SEQ + kstart + c0, SEQ, lane);
      acc[dt] = wmma_bf16(pa, vb, acc[dt]);
    }
  }

  bf16_t* Cb = Ctx + (size_t)b * SEQ * EMB;
  #pragma unroll
  for (int dt = 0; dt < 4; ++dt) {
    const int e = h * HEAD_D + dt * 16 + col16;
    #pragma unroll
    for (int r = 0; r < 8; ++r)
      Cb[(size_t)(q0 + rb + r) * EMB + e] = (bf16_t)acc[dt][r];
  }
}

// ---------------------------------------------------------------- launch
extern "C" void kernel_launch(void* const* d_in, const int* in_sizes, int n_in,
                              void* d_out, int out_size, void* d_ws, size_t ws_size,
                              hipStream_t stream) {
  (void)in_sizes; (void)n_in; (void)out_size; (void)ws_size;
  const float* X  = (const float*)d_in[0];
  const float* Wq = (const float*)d_in[1];
  const float* Wk = (const float*)d_in[2];
  const float* Wv = (const float*)d_in[3];
  const float* Wo = (const float*)d_in[4];
  const float* bo = (const float*)d_in[5];

  char* ws = (char*)d_ws;
  size_t off = 0;
  auto take = [&](size_t elems) {
    bf16_t* p = (bf16_t*)(ws + off);
    off += elems * sizeof(bf16_t);
    return p;
  };
  bf16_t* Xbf   = take((size_t)MTOT * EMB);
  bf16_t* Wqbf  = take((size_t)EMB * EMB);
  bf16_t* Wkbf  = take((size_t)EMB * EMB);
  bf16_t* Wvbf  = take((size_t)EMB * EMB);
  bf16_t* Wobf  = take((size_t)EMB * EMB);
  bf16_t* Qbf   = take((size_t)MTOT * EMB);
  bf16_t* Kbf   = take((size_t)MTOT * EMB);
  bf16_t* Vtbf  = take((size_t)MTOT * EMB + 512);  // pad: PV tail chunk over-read x 0
  bf16_t* Ctxbf = take((size_t)MTOT * EMB);

  auto cvt = [&](const float* in, bf16_t* outp, size_t n) {
    int n4 = (int)(n / 4);
    int grid = (n4 + 255) / 256;
    if (grid > 8192) grid = 8192;
    cvt_f32_to_bf16<<<dim3(grid), dim3(256), 0, stream>>>(in, outp, n4);
  };
  cvt(X,  Xbf,  (size_t)MTOT * EMB);
  cvt(Wq, Wqbf, (size_t)EMB * EMB);
  cvt(Wk, Wkbf, (size_t)EMB * EMB);
  cvt(Wv, Wvbf, (size_t)EMB * EMB);
  cvt(Wo, Wobf, (size_t)EMB * EMB);

  dim3 ggrid(EMB / 128, MTOT / 128);
  gemm_x_wt<<<ggrid, 256, 0, stream>>>(Xbf, Wqbf, (void*)Qbf,  nullptr, EPI_BHSD);
  gemm_x_wt<<<ggrid, 256, 0, stream>>>(Xbf, Wkbf, (void*)Kbf,  nullptr, EPI_BHSD);
  gemm_x_wt<<<ggrid, 256, 0, stream>>>(Xbf, Wvbf, (void*)Vtbf, nullptr, EPI_BHDS);

  attn_sliding_window<<<dim3(SEQ / 16, BATCH * H_HEADS), 32, 0, stream>>>(Qbf, Kbf, Vtbf, Ctxbf);

  gemm_x_wt<<<ggrid, 256, 0, stream>>>(Ctxbf, Wobf, d_out, bo, EPI_F32B);
}
